// Cross_Modal_Attention_Fusion_53798760350103
// MI455X (gfx1250) — compile-verified, matrix operand order checked by reading, not run on hardware
//
#include <hip/hip_runtime.h>

// ---------------------------------------------------------------------------
// CDNA5 (gfx1250) implementation of Cross_Modal_Attention_Fusion.
// All large einsums -> tiled bf16 WMMA GEMM (v_wmma_f32_16x16x32_bf16),
// fp32 accumulate, software-pipelined global->LDS staging with vectorized
// b128 loads/stores. Glue kernels use deterministic two-pass reductions.
// ---------------------------------------------------------------------------

typedef __attribute__((ext_vector_type(16))) __bf16 v16bf;
typedef __attribute__((ext_vector_type(8)))  __bf16 v8bf;
typedef __attribute__((ext_vector_type(8)))  float  v8f;

// Problem constants
#define BB   16
#define LL   512
#define DD   1024
#define HH   1024
#define MM   512
#define LH   (LL * HH)           // 524288
#define BLH  (BB * LL * HH)      // 8388608
#define HHH  (HH * HH)           // 1048576
#define POOL_N 917504            // 7*256*512
#define FEAT_ROWS 1792           // 7*256

// GEMM tiling
#define BLK_M 64
#define BLK_N 128
#define BLK_K 32
#define LDSK  40                 // padded k-stride (bf16 elems): 80B rows, keeps
                                 // 16B alignment for packed v8bf stores

static __device__ __forceinline__ float4 ld4(const float* p) {
  return *(const float4*)p;
}

// ---------------------------------------------------------------------------
// Generic tiled WMMA GEMM:  C[M,N] = A ? B (+bias), fp32 in/out, bf16 compute.
//   TA=false: A stored row-major [M,K];  TA=true: A stored [K,M] (A^T applied)
//   TB=false: B stored row-major [K,N];  TB=true: B stored [N,K] (B^T applied)
// Batched via blockIdx.z with element strides (0 => shared operand).
// Requires M%64==0, N%128==0, K%32==0 (true for every call site here).
// ---------------------------------------------------------------------------
template <bool TA, bool TB, bool BIAS>
__global__ __launch_bounds__(256) void gemm_wmma_bf16(
    const float* __restrict__ A, const float* __restrict__ B,
    const float* __restrict__ bias, float* __restrict__ C,
    int M, int N, int K,
    long long strideA, long long strideB, long long strideC)
{
  __shared__ __bf16 As[BLK_M][LDSK];
  __shared__ __bf16 Bs[BLK_N][LDSK];

  const int tid  = threadIdx.x;
  const int lane = tid & 31;
  const int wave = tid >> 5;
  const int mw   = (wave & 1) * 32;   // wave grid: 2 (M) x 4 (N)
  const int nw   = (wave >> 1) * 32;

  const int m0 = blockIdx.x * BLK_M;
  const int n0 = blockIdx.y * BLK_N;
  A += (long long)blockIdx.z * strideA;
  B += (long long)blockIdx.z * strideB;
  C += (long long)blockIdx.z * strideC;

  // ---- per-thread staging geometry (computed once) ----
  // A tile: 64x32 = 2048 elems, 8 contiguous fp32 per thread.
  // B tile: 128x32 = 4096 elems, 16 contiguous fp32 per thread.
  const int aM  = tid >> 2, aK8  = (tid & 3) * 8;   // TA=false: row aM, k offs aK8
  const int aK  = tid >> 3, aM8  = (tid & 7) * 8;   // TA=true : k row aK, m offs aM8
  const int bK  = tid >> 3, bN16 = (tid & 7) * 16;  // TB=false: k row bK, n offs bN16
  const int bN  = tid >> 1, bK16 = (tid & 1) * 16;  // TB=true : row bN, k offs bK16

  const float* pa = TA ? (A + (long long)aK * M + (m0 + aM8))
                       : (A + (long long)(m0 + aM) * K + aK8);
  const float* pb = TB ? (B + (long long)(n0 + bN) * K + bK16)
                       : (B + (long long)bK * N + (n0 + bN16));
  const long long stepA = TA ? (long long)BLK_K * M : (long long)BLK_K;
  const long long stepB = TB ? (long long)BLK_K     : (long long)BLK_K * N;

  float4 ra0, ra1, rb0, rb1, rb2, rb3;

  auto loadAB = [&]() {
    ra0 = ld4(pa); ra1 = ld4(pa + 4);
    rb0 = ld4(pb); rb1 = ld4(pb + 4); rb2 = ld4(pb + 8); rb3 = ld4(pb + 12);
    pa += stepA; pb += stepB;
  };

  auto storeAB = [&]() {
    if (!TA) {                       // contiguous k: packed 8x bf16 store (b128)
      v8bf t;
      t[0] = (__bf16)ra0.x; t[1] = (__bf16)ra0.y; t[2] = (__bf16)ra0.z; t[3] = (__bf16)ra0.w;
      t[4] = (__bf16)ra1.x; t[5] = (__bf16)ra1.y; t[6] = (__bf16)ra1.z; t[7] = (__bf16)ra1.w;
      *reinterpret_cast<v8bf*>(&As[aM][aK8]) = t;
    } else {                         // transposed: 8 scalar b16 stores, strided rows
      As[aM8 + 0][aK] = (__bf16)ra0.x; As[aM8 + 1][aK] = (__bf16)ra0.y;
      As[aM8 + 2][aK] = (__bf16)ra0.z; As[aM8 + 3][aK] = (__bf16)ra0.w;
      As[aM8 + 4][aK] = (__bf16)ra1.x; As[aM8 + 5][aK] = (__bf16)ra1.y;
      As[aM8 + 6][aK] = (__bf16)ra1.z; As[aM8 + 7][aK] = (__bf16)ra1.w;
    }
    if (!TB) {                       // n-contiguous in global, k-major in LDS
      Bs[bN16 +  0][bK] = (__bf16)rb0.x; Bs[bN16 +  1][bK] = (__bf16)rb0.y;
      Bs[bN16 +  2][bK] = (__bf16)rb0.z; Bs[bN16 +  3][bK] = (__bf16)rb0.w;
      Bs[bN16 +  4][bK] = (__bf16)rb1.x; Bs[bN16 +  5][bK] = (__bf16)rb1.y;
      Bs[bN16 +  6][bK] = (__bf16)rb1.z; Bs[bN16 +  7][bK] = (__bf16)rb1.w;
      Bs[bN16 +  8][bK] = (__bf16)rb2.x; Bs[bN16 +  9][bK] = (__bf16)rb2.y;
      Bs[bN16 + 10][bK] = (__bf16)rb2.z; Bs[bN16 + 11][bK] = (__bf16)rb2.w;
      Bs[bN16 + 12][bK] = (__bf16)rb3.x; Bs[bN16 + 13][bK] = (__bf16)rb3.y;
      Bs[bN16 + 14][bK] = (__bf16)rb3.z; Bs[bN16 + 15][bK] = (__bf16)rb3.w;
    } else {                         // contiguous k: two packed b128 stores
      v8bf t0, t1;
      t0[0] = (__bf16)rb0.x; t0[1] = (__bf16)rb0.y; t0[2] = (__bf16)rb0.z; t0[3] = (__bf16)rb0.w;
      t0[4] = (__bf16)rb1.x; t0[5] = (__bf16)rb1.y; t0[6] = (__bf16)rb1.z; t0[7] = (__bf16)rb1.w;
      t1[0] = (__bf16)rb2.x; t1[1] = (__bf16)rb2.y; t1[2] = (__bf16)rb2.z; t1[3] = (__bf16)rb2.w;
      t1[4] = (__bf16)rb3.x; t1[5] = (__bf16)rb3.y; t1[6] = (__bf16)rb3.z; t1[7] = (__bf16)rb3.w;
      *reinterpret_cast<v8bf*>(&Bs[bN][bK16])     = t0;
      *reinterpret_cast<v8bf*>(&Bs[bN][bK16 + 8]) = t1;
    }
  };

  v8f acc[2][2] = {};

  loadAB();                                       // tile 0 -> registers
  for (int k0 = 0; k0 < K; k0 += BLK_K) {
    storeAB();                                    // registers -> LDS (cvt + ds)
    __syncthreads();

    if (k0 + BLK_K < K) {
      loadAB();                                   // issue next-tile loads; the
                                                  // wait lands in next storeAB,
                                                  // overlapping with WMMA below
      if (k0 + 2 * BLK_K < K) {                   // prefetch 2 tiles ahead
        __builtin_prefetch((const void*)pa, 0, 1);
        __builtin_prefetch((const void*)pb, 0, 1);
      }
    }

    // ---- per-lane fragment loads (16-bit 16x32 A / 32x16 B layout) ----
    // lanes 0-15: K = {kb..kb+7, kb+16..kb+23} with kb=0 ; lanes 16-31: kb=8
    const int kb = (lane & 16) ? 8 : 0;
    const int rc = lane & 15;
    v16bf fa[2], fb[2];
#pragma unroll
    for (int i = 0; i < 2; ++i) {
      const __bf16* p = &As[mw + 16 * i + rc][0];
#pragma unroll
      for (int e = 0; e < 8; ++e) { fa[i][e] = p[kb + e]; fa[i][8 + e] = p[kb + 16 + e]; }
    }
#pragma unroll
    for (int j = 0; j < 2; ++j) {
      const __bf16* p = &Bs[nw + 16 * j + rc][0];
#pragma unroll
      for (int e = 0; e < 8; ++e) { fb[j][e] = p[kb + e]; fb[j][8 + e] = p[kb + 16 + e]; }
    }

#pragma unroll
    for (int i = 0; i < 2; ++i)
#pragma unroll
      for (int j = 0; j < 2; ++j)
        acc[i][j] = __builtin_amdgcn_wmma_f32_16x16x32_bf16(
            false, fa[i], false, fb[j], (short)0, acc[i][j], false, false);

    __syncthreads();
  }

  // ---- epilogue: C/D layout = lane n = lane&15, rows r + (lane>=16 ? 8 : 0)
  const int rc    = lane & 15;
  const int rbase = (lane & 16) ? 8 : 0;
#pragma unroll
  for (int i = 0; i < 2; ++i)
#pragma unroll
    for (int j = 0; j < 2; ++j) {
      const int col = n0 + nw + 16 * j + rc;
      const float bv = BIAS ? bias[col] : 0.0f;
#pragma unroll
      for (int r = 0; r < 8; ++r) {
        const int row = m0 + mw + 16 * i + rbase + r;
        C[(long long)row * N + col] = acc[i][j][r] + bv;
      }
    }
}

// ---------------------------------------------------------------------------
// Reductions (deterministic two-pass)
// ---------------------------------------------------------------------------
__global__ __launch_bounds__(256) void exp_sum_partial(
    const float* __restrict__ in, long long n, float* __restrict__ partial)
{
  __shared__ float red[256];
  float s = 0.0f;
  for (long long i = blockIdx.x * 256LL + threadIdx.x; i < n; i += (long long)gridDim.x * 256LL)
    s += expf(in[i]);
  red[threadIdx.x] = s; __syncthreads();
  for (int t = 128; t > 0; t >>= 1) {
    if (threadIdx.x < t) red[threadIdx.x] += red[threadIdx.x + t];
    __syncthreads();
  }
  if (threadIdx.x == 0) partial[blockIdx.x] = red[0];
}

__global__ __launch_bounds__(256) void exp_inplace_partial(
    float* __restrict__ buf, long long n, float* __restrict__ partial)
{
  __shared__ float red[256];
  float s = 0.0f;
  for (long long i = blockIdx.x * 256LL + threadIdx.x; i < n; i += (long long)gridDim.x * 256LL) {
    float e = expf(buf[i]);
    buf[i] = e;
    s += e;
  }
  red[threadIdx.x] = s; __syncthreads();
  for (int t = 128; t > 0; t >>= 1) {
    if (threadIdx.x < t) red[threadIdx.x] += red[threadIdx.x + t];
    __syncthreads();
  }
  if (threadIdx.x == 0) partial[blockIdx.x] = red[0];
}

__global__ __launch_bounds__(256) void reduce_partials(
    const float* __restrict__ partial, int nb, float* __restrict__ out)
{
  __shared__ float red[256];
  float s = 0.0f;
  for (int i = threadIdx.x; i < nb; i += 256) s += partial[i];
  red[threadIdx.x] = s; __syncthreads();
  for (int t = 128; t > 0; t >>= 1) {
    if (threadIdx.x < t) red[threadIdx.x] += red[threadIdx.x + t];
    __syncthreads();
  }
  if (threadIdx.x == 0) out[0] = red[0];
}

__global__ __launch_bounds__(256) void scale_inplace(
    float* __restrict__ buf, const float* __restrict__ sum, long long n)
{
  const float inv = 1.0f / sum[0];
  for (long long i = blockIdx.x * 256LL + threadIdx.x; i < n; i += (long long)gridDim.x * 256LL)
    buf[i] *= inv;
}

// t[i] *= exp(gate[i & mask]) / sum[0]   (gate map broadcast over batch)
__global__ __launch_bounds__(256) void gate_mul(
    float* __restrict__ t, const float* __restrict__ gate,
    const float* __restrict__ sum, long long n, long long mask)
{
  const float inv = 1.0f / sum[0];
  for (long long i = blockIdx.x * 256LL + threadIdx.x; i < n; i += (long long)gridDim.x * 256LL)
    t[i] *= expf(gate[i & mask]) * inv;
}

__global__ __launch_bounds__(256) void add_inplace(
    float* __restrict__ a, const float* __restrict__ b, long long n)
{
  long long i = blockIdx.x * 256LL + threadIdx.x;
  if (i < n) a[i] += b[i];
}

// ---------------------------------------------------------------------------
// SoftPool3D over (4,2,2) windows, stride (2,2,2), VALID. in: [16,512,1024] ->
// out: [7,256,512].  out = sum(e*v)/sum(e) (+ optional addin).
// ---------------------------------------------------------------------------
__global__ __launch_bounds__(256) void softpool3d_k(
    const float* __restrict__ in, const float* __restrict__ addin,
    float* __restrict__ out, int D1, int D2, int O1, int O2, long long total)
{
  long long idx = blockIdx.x * 256LL + threadIdx.x;
  if (idx >= total) return;
  int k = (int)(idx % O2);
  int j = (int)((idx / O2) % O1);
  int i = (int)(idx / ((long long)O2 * O1));
  float num = 0.0f, den = 0.0f;
#pragma unroll
  for (int db = 0; db < 4; ++db)
#pragma unroll
    for (int dl = 0; dl < 2; ++dl) {
      const float* row = &in[((long long)(2 * i + db) * D1 + (2 * j + dl)) * D2 + 2 * k];
#pragma unroll
      for (int dd = 0; dd < 2; ++dd) {
        float v = row[dd];
        float e = expf(v);
        num += e * v;
        den += e;
      }
    }
  float r = num / den;
  if (addin) r += addin[idx];
  out[idx] = r;
}

// Row softmax for N=512 (one block of 256 per row, 2 elems/thread)
__global__ __launch_bounds__(256) void softmax512(
    const float* __restrict__ in, float* __restrict__ out)
{
  __shared__ float red[256];
  const long long base = (long long)blockIdx.x * 512;
  float a = in[base + threadIdx.x];
  float b = in[base + threadIdx.x + 256];
  red[threadIdx.x] = fmaxf(a, b); __syncthreads();
  for (int t = 128; t > 0; t >>= 1) {
    if (threadIdx.x < t) red[threadIdx.x] = fmaxf(red[threadIdx.x], red[threadIdx.x + t]);
    __syncthreads();
  }
  const float m = red[0]; __syncthreads();
  float ea = expf(a - m), eb = expf(b - m);
  red[threadIdx.x] = ea + eb; __syncthreads();
  for (int t = 128; t > 0; t >>= 1) {
    if (threadIdx.x < t) red[threadIdx.x] += red[threadIdx.x + t];
    __syncthreads();
  }
  const float inv = 1.0f / red[0];
  out[base + threadIdx.x]       = ea * inv;
  out[base + threadIdx.x + 256] = eb * inv;
}

// feature = concat(w1 * x_hat, w2 * y_hat) along last dim (512+512 -> 1024)
__global__ __launch_bounds__(256) void concat_scale(
    const float* __restrict__ xs, const float* __restrict__ ys,
    const float* __restrict__ w, float* __restrict__ feat, long long total)
{
  long long idx = blockIdx.x * 256LL + threadIdx.x;
  if (idx >= total) return;
  const float e0 = expf(w[0]), e1 = expf(w[1]);
  const float inv = 1.0f / (e0 + e1);
  long long r = idx >> 10;
  int c = (int)(idx & 1023);
  feat[idx] = (c < 512) ? (e0 * inv) * xs[r * 512 + c]
                        : (e1 * inv) * ys[r * 512 + (c - 512)];
}

// ---------------------------------------------------------------------------
// Host-side orchestration
// ---------------------------------------------------------------------------
extern "C" void kernel_launch(void* const* d_in, const int* in_sizes, int n_in,
                              void* d_out, int out_size, void* d_ws, size_t ws_size,
                              hipStream_t stream) {
  const float* x       = (const float*)d_in[0];
  const float* y       = (const float*)d_in[1];
  const float* w_fc1_t = (const float*)d_in[2];
  const float* b_fc1_t = (const float*)d_in[3];
  const float* w_fc1_v = (const float*)d_in[4];
  const float* b_fc1_v = (const float*)d_in[5];
  const float* wq_t    = (const float*)d_in[6];
  const float* wk_t    = (const float*)d_in[7];
  const float* wv_t    = (const float*)d_in[8];
  const float* wq_v    = (const float*)d_in[9];
  const float* wk_v    = (const float*)d_in[10];
  const float* wv_v    = (const float*)d_in[11];
  const float* w_fc2_t = (const float*)d_in[12];
  const float* b_fc2_t = (const float*)d_in[13];
  const float* w_fc2_v = (const float*)d_in[14];
  const float* b_fc2_v = (const float*)d_in[15];
  const float* w_t     = (const float*)d_in[16];
  const float* w_v     = (const float*)d_in[17];
  const float* w_pair  = (const float*)d_in[18];

  float* ws = (float*)d_ws;
  const long long E = BLH;                       // 8,388,608 floats
  float* XH  = ws;                               // fc1(x) -> later reused for ox
  float* YH  = ws + 1 * E;                       // fc1(y) -> later reused for oy
  float* QT  = ws + 2 * E;
  float* KT  = ws + 3 * E;
  float* VT  = ws + 4 * E;
  float* QV  = ws + 5 * E;
  float* KV  = ws + 6 * E;
  float* VV  = ws + 7 * E;
  float* AX  = ws + 8 * E;                       // [16,1024,1024]  (2E)
  float* AY  = ws + 10 * E;                      // [16,1024,1024]  (2E)
  float* base = ws + 12 * E;
  float* XSP = base;            base += POOL_N;  // softpool(x)
  float* YSP = base;            base += POOL_N;
  float* XHP = base;            base += POOL_N;  // softpool(ox)+XSP
  float* YHP = base;            base += POOL_N;
  float* F2X = base;            base += POOL_N;  // fc2 outputs
  float* F2Y = base;            base += POOL_N;
  float* PART = base;           base += 1024;    // partial sums
  float* SC   = base;           /* 8 scalars */  // [0]=sum et, [1]=sum ev, [2]=sum ax, [3]=sum ay

  float* out      = (float*)d_out;
  float* out_feat = out;                                         // [7,256,1024]
  float* out_xh   = out + (long long)FEAT_ROWS * 1024;           // [7,256,512]
  float* out_yh   = out_xh + (long long)FEAT_ROWS * 512;         // [7,256,512]

  const dim3 blk(256);

  // ---- gate-map normalizers: sum(exp(w_t)), sum(exp(w_v)) over 3*L*H ----
  exp_sum_partial<<<256, blk, 0, stream>>>(w_t, 3LL * LH, PART);
  reduce_partials<<<1, blk, 0, stream>>>(PART, 256, SC + 0);
  exp_sum_partial<<<256, blk, 0, stream>>>(w_v, 3LL * LH, PART);
  reduce_partials<<<1, blk, 0, stream>>>(PART, 256, SC + 1);

  // ---- fc1: [8192,1024] @ [1024,1024]^T + bias ----
  {
    dim3 g(8192 / BLK_M, HH / BLK_N, 1);
    gemm_wmma_bf16<false, true, true><<<g, blk, 0, stream>>>(
        x, w_fc1_t, b_fc1_t, XH, 8192, HH, DD, 0, 0, 0);
    gemm_wmma_bf16<false, true, true><<<g, blk, 0, stream>>>(
        y, w_fc1_v, b_fc1_v, YH, 8192, HH, DD, 0, 0, 0);
  }

  // ---- conv1x1 projections: per-batch  W[512,512] @ xh_b[512,1024] ----
  {
    dim3 g(LL / BLK_M, HH / BLK_N, BB);
    gemm_wmma_bf16<false, false, false><<<g, blk, 0, stream>>>(wq_t, XH, nullptr, QT, LL, HH, LL, 0, LH, LH);
    gemm_wmma_bf16<false, false, false><<<g, blk, 0, stream>>>(wk_t, XH, nullptr, KT, LL, HH, LL, 0, LH, LH);
    gemm_wmma_bf16<false, false, false><<<g, blk, 0, stream>>>(wv_t, XH, nullptr, VT, LL, HH, LL, 0, LH, LH);
    gemm_wmma_bf16<false, false, false><<<g, blk, 0, stream>>>(wq_v, YH, nullptr, QV, LL, HH, LL, 0, LH, LH);
    gemm_wmma_bf16<false, false, false><<<g, blk, 0, stream>>>(wk_v, YH, nullptr, KV, LL, HH, LL, 0, LH, LH);
    gemm_wmma_bf16<false, false, false><<<g, blk, 0, stream>>>(wv_v, YH, nullptr, VV, LL, HH, LL, 0, LH, LH);
  }

  // ---- apply normalized exp gates in place ----
  {
    const int gb = 1024;
    gate_mul<<<gb, blk, 0, stream>>>(QT, w_t + 0LL * LH, SC + 0, BLH, LH - 1);  // wt_q * q_t
    gate_mul<<<gb, blk, 0, stream>>>(KT, w_t + 1LL * LH, SC + 0, BLH, LH - 1);  // wt_k * k_t
    gate_mul<<<gb, blk, 0, stream>>>(VT, w_t + 2LL * LH, SC + 0, BLH, LH - 1);  // wt_val * v_t
    gate_mul<<<gb, blk, 0, stream>>>(QV, w_v + 0LL * LH, SC + 1, BLH, LH - 1);  // wv_q * q_v
    gate_mul<<<gb, blk, 0, stream>>>(KV, w_v + 1LL * LH, SC + 1, BLH, LH - 1);  // wv_k * k_v
    gate_mul<<<gb, blk, 0, stream>>>(VV, w_v + 2LL * LH, SC + 1, BLH, LH - 1);  // wv_val * v_v
  }

  // ---- attention logits (A^T B over l), exp, global-sum normalize ----
  {
    dim3 g(HH / BLK_M, HH / BLK_N, BB);
    gemm_wmma_bf16<true, false, false><<<g, blk, 0, stream>>>(KV, QT, nullptr, AX, HH, HH, LL, LH, LH, HHH);
    gemm_wmma_bf16<true, false, false><<<g, blk, 0, stream>>>(KT, QV, nullptr, AY, HH, HH, LL, LH, LH, HHH);
  }
  exp_inplace_partial<<<1024, blk, 0, stream>>>(AX, (long long)BB * HHH, PART);
  reduce_partials<<<1, blk, 0, stream>>>(PART, 1024, SC + 2);
  scale_inplace<<<1024, blk, 0, stream>>>(AX, SC + 2, (long long)BB * HHH);
  exp_inplace_partial<<<1024, blk, 0, stream>>>(AY, (long long)BB * HHH, PART);
  reduce_partials<<<1, blk, 0, stream>>>(PART, 1024, SC + 3);
  scale_inplace<<<1024, blk, 0, stream>>>(AY, SC + 3, (long long)BB * HHH);

  // ---- ox = (gated v_v) @ a_x ; oy = (gated v_t) @ a_y (reuse XH/YH) ----
  {
    dim3 g(LL / BLK_M, HH / BLK_N, BB);
    gemm_wmma_bf16<false, false, false><<<g, blk, 0, stream>>>(VV, AX, nullptr, XH, LL, HH, HH, LH, HHH, LH);
    gemm_wmma_bf16<false, false, false><<<g, blk, 0, stream>>>(VT, AY, nullptr, YH, LL, HH, HH, LH, HHH, LH);
  }

  // ---- softpool3d stage ----
  {
    const int gb = (POOL_N + 255) / 256;
    softpool3d_k<<<gb, blk, 0, stream>>>(x,  nullptr, XSP, LL, DD, 256, 512, POOL_N);
    softpool3d_k<<<gb, blk, 0, stream>>>(y,  nullptr, YSP, LL, DD, 256, 512, POOL_N);
    softpool3d_k<<<gb, blk, 0, stream>>>(XH, XSP,     XHP, LL, HH, 256, 512, POOL_N);  // softpool(ox)+x_sp
    softpool3d_k<<<gb, blk, 0, stream>>>(YH, YSP,     YHP, LL, HH, 256, 512, POOL_N);  // softpool(oy)+y_sp
  }

  // ---- fc2 (crossed weights) ----
  {
    dim3 g(FEAT_ROWS / BLK_M, MM / BLK_N, 1);
    gemm_wmma_bf16<false, true, true><<<g, blk, 0, stream>>>(
        XHP, w_fc2_v, b_fc2_v, F2X, FEAT_ROWS, MM, HH / 2, 0, 0, 0);
    gemm_wmma_bf16<false, true, true><<<g, blk, 0, stream>>>(
        YHP, w_fc2_t, b_fc2_t, F2Y, FEAT_ROWS, MM, HH / 2, 0, 0, 0);
  }

  // ---- residual on x path, then row softmax into output slots ----
  add_inplace<<<(POOL_N + 255) / 256, blk, 0, stream>>>(F2X, XSP, POOL_N);
  softmax512<<<FEAT_ROWS, blk, 0, stream>>>(F2X, out_xh);
  softmax512<<<FEAT_ROWS, blk, 0, stream>>>(F2Y, out_yh);

  // ---- feature = concat(w1 * x_hat, w2 * y_hat) ----
  {
    const long long tot = (long long)FEAT_ROWS * 1024;
    concat_scale<<<(int)((tot + 255) / 256), blk, 0, stream>>>(out_xh, out_yh, w_pair, out_feat, tot);
  }
}
